// roi_grid_head_19937238188280
// MI455X (gfx1250) — matrix operand order repaired and checked by reading, not compiled, not run on hardware
//
#include <hip/hip_runtime.h>

#define NROI   512
#define DIMIN  256
#define COUT   576
#define P      9
#define CPT    64
#define NE     24
#define HH     7
#define WW     7
#define HW     49
#define GROUPS 36

typedef __attribute__((ext_vector_type(16))) _Float16 v16h;
typedef __attribute__((ext_vector_type(8)))  _Float16 v8h;
typedef __attribute__((ext_vector_type(8)))  float    v8f;

// 3x3 grid edges (same construction order as reference)
__constant__ int d_SRC[NE] = {1,3, 0,2,4, 1,5, 0,4,6, 1,3,5,7, 2,4,8, 3,7, 4,6,8, 5,7};
__constant__ int d_DST[NE] = {0,0, 1,1,1, 2,2, 3,3,3, 4,4,4,4, 5,5,5, 6,6, 7,7,7, 8,8};

__device__ __forceinline__ v16h join16(v8h lo, v8h hi) {
  return __builtin_shufflevector(lo, hi, 0, 1, 2, 3, 4, 5, 6, 7,
                                 8, 9, 10, 11, 12, 13, 14, 15);
}

// ---------------------------------------------------------------------------
// Weight prepack: w[co][ci][ky][kx] (f32) -> wp[co][tap][ci] (f16), tap=ky*3+kx
// Makes B-fragment loads contiguous in K (=ci) for WMMA.
// ---------------------------------------------------------------------------
__global__ void pack_conv_w(const float* __restrict__ w, _Float16* __restrict__ wp,
                            int CO, int CI) {
  const size_t total = (size_t)CO * CI * 9;
  size_t i = (size_t)blockIdx.x * 256 + threadIdx.x;
  if (i >= total) return;
  const int    ci  = (int)(i % CI);
  const size_t t   = i / CI;
  const int    tap = (int)(t % 9);
  const int    co  = (int)(t / 9);
  wp[i] = (_Float16)w[((size_t)co * CI + ci) * 9 + tap];
}

__global__ void cvt_f32_to_f16(const float* __restrict__ s, _Float16* __restrict__ d, int n) {
  const int i = blockIdx.x * 256 + threadIdx.x;
  if (i < n) d[i] = (_Float16)s[i];
}

// ---------------------------------------------------------------------------
// Implicit-GEMM 3x3 conv with WMMA f16->f32.
// grid = (NROI, COUT/32), block = 256 (8 waves).
// Whole per-ROI input patch staged in LDS as f16, 1-px zero halo, TRANSPOSED
// to [pixel][ci] so A-fragment K-runs are contiguous -> 2x ds_load_b128 per
// WMMA. (576*81 = 93KB or 256*256 = 131KB f16; CDNA5 WGP has 320KB LDS.)
// GEMM: M = 49(pad 64) output pixels, N = 32 output channels, K = CIN*9.
// Wave (mi,nj) owns one 16x16 tile; K stepped by 32 (one WMMA per step).
// ---------------------------------------------------------------------------
__global__ void conv3x3_gemm_wmma(const float* __restrict__ in,
                                  const _Float16* __restrict__ wp,
                                  const float* __restrict__ bias,
                                  float* __restrict__ out,
                                  int CIN, int IH, int IW, int stride) {
  extern __shared__ _Float16 patch[];  // [(IH+2)*(IW+2)][CIN]
  const int IWP = IW + 2;
  const int PP  = (IH + 2) * IWP;
  const int n   = blockIdx.x;
  const int tid = threadIdx.x;

  // Stage: coalesced global reads (pixel-fast), transposed LDS writes.
  const float* inb = in + (size_t)n * CIN * IH * IW;
  for (int idx = tid; idx < CIN * PP; idx += 256) {
    const int ci = idx / PP;
    const int r  = idx - ci * PP;
    const int py = r / IWP;
    const int px = r - py * IWP;
    const int iy = py - 1, ix = px - 1;
    float v = 0.0f;
    if (iy >= 0 && iy < IH && ix >= 0 && ix < IW)
      v = inb[(size_t)ci * (IH * IW) + iy * IW + ix];
    patch[(size_t)r * CIN + ci] = (_Float16)v;
  }
  __syncthreads();

  const int wave = tid >> 5;
  const int lane = tid & 31;
  const int half = lane >> 4;   // lane group: selects K sub-base
  const int l15  = lane & 15;
  const int mi   = wave >> 1;   // 0..3 : m-subtile
  const int nj   = wave & 1;    // 0..1 : n-subtile

  int m = mi * 16 + l15;
  if (m > 48) m = 48;           // pad rows duplicate last pixel; stores masked
  const int oy  = m / WW;
  const int ox  = m - oy * WW;
  const int co  = blockIdx.y * 32 + nj * 16 + l15;
  const int kbA = half * 16;    // A: K base per lane-half (elems 0..15 contiguous)
  const int kbB = half * 16;    // B: K base per lane-half

  v8f acc = {};
  for (int tap = 0; tap < 9; ++tap) {
    const int dy = tap / 3;
    const int dx = tap - dy * 3;
    const int pix = (oy * stride + dy) * IWP + (ox * stride + dx);
    const _Float16* wrow = wp + ((size_t)co * 9 + tap) * CIN + kbB;
    const _Float16* arow = patch + (size_t)pix * CIN + kbA;
    if (tap < 8) __builtin_prefetch(wrow + CIN, 0, 1);  // next tap's B row -> L2
    for (int ci0 = 0; ci0 < CIN; ci0 += 32) {
      const v8h alo = *(const v8h*)(arow + ci0);        // K = ci0+kbA .. +7
      const v8h ahi = *(const v8h*)(arow + ci0 + 8);    // K = ci0+kbA+8 .. +15
      const v16h a  = join16(alo, ahi);
      const v16h b  = *(const v16h*)(wrow + ci0);       // contiguous in K
      acc = __builtin_amdgcn_wmma_f32_16x16x32_f16(false, a, false, b,
                                                   (short)0, acc, false, false);
    }
  }

  const float bv = bias[co];
  float* ob = out + ((size_t)n * COUT + co) * HW;
#pragma unroll
  for (int v = 0; v < 8; ++v) {
    const int mr = mi * 16 + half * 8 + v;   // C/D row mapping
    if (mr < HW) ob[mr] = acc[v] + bv;
  }
}

// ---------------------------------------------------------------------------
// GroupNorm(36 groups of 16 ch) + ReLU, in place. grid=(NROI,36), block=256.
// ---------------------------------------------------------------------------
__global__ void gn_relu(float* __restrict__ xbuf, const float* __restrict__ sc,
                        const float* __restrict__ sh) {
  __shared__ float rs[256], rq[256];
  const int n = blockIdx.x, g = blockIdx.y;
  float* base = xbuf + ((size_t)n * COUT + g * 16) * HW;
  const int tid = threadIdx.x;
  float s0 = 0.f, s1 = 0.f;
  for (int i = tid; i < 16 * HW; i += 256) {
    const float v = base[i];
    s0 += v; s1 += v * v;
  }
  rs[tid] = s0; rq[tid] = s1;
  __syncthreads();
  for (int st = 128; st > 0; st >>= 1) {
    if (tid < st) { rs[tid] += rs[tid + st]; rq[tid] += rq[tid + st]; }
    __syncthreads();
  }
  const float inv_n = 1.0f / (16.0f * HW);
  const float mean  = rs[0] * inv_n;
  const float var   = rq[0] * inv_n - mean * mean;
  const float rinv  = rsqrtf(var + 1e-5f);
  for (int i = tid; i < 16 * HW; i += 256) {
    const int ch = g * 16 + i / HW;
    const float v = (base[i] - mean) * rinv * sc[ch] + sh[ch];
    base[i] = v > 0.f ? v : 0.f;
  }
}

// ---------------------------------------------------------------------------
// Per-(ROI, edge) message: stage src point, depthwise 5x5 (pad 2) into LDS f16
// A-matrix [49->64 x K=64] (K contiguous -> 2x ds_load_b128 per WMMA), then
// WMMA against prepacked f16 pointwise weights (B[k=i][col=o] = pw[e][o][i],
// contiguous in i), atomic scatter-add to the destination point.
// grid=(NROI, NE), block=256 (8 waves, 16 WMMA tiles -> 2 per wave).
// ---------------------------------------------------------------------------
__global__ void msg_dwpw_wmma(const float* __restrict__ srcfeat,
                              float* __restrict__ dstfeat,
                              const float* __restrict__ dw_w,
                              const float* __restrict__ dw_b,
                              const _Float16* __restrict__ pw,
                              const float* __restrict__ pw_b) {
  __shared__ float    gbuf[CPT * HW];
  __shared__ _Float16 As[64 * 64];
  const int n = blockIdx.x, e = blockIdx.y;
  const int spt = d_SRC[e], dpt = d_DST[e];
  const int tid = threadIdx.x;

  const float* src = srcfeat + ((size_t)n * P + spt) * CPT * HW;
  for (int i = tid; i < CPT * HW; i += 256) gbuf[i] = src[i];
  __syncthreads();

  for (int i = tid; i < CPT * HW; i += 256) {
    const int c = i / HW;
    const int r = i - c * HW;
    const int y = r / WW, x = r - y * WW;
    const float* wd = dw_w + ((size_t)e * CPT + c) * 25;
    const float* gc = gbuf + c * HW;
    float acc = dw_b[e * CPT + c];
#pragma unroll
    for (int ky = 0; ky < 5; ++ky) {
      const int iy = y + ky - 2;
      if (iy < 0 || iy >= HH) continue;
#pragma unroll
      for (int kx = 0; kx < 5; ++kx) {
        const int ix = x + kx - 2;
        if (ix < 0 || ix >= WW) continue;
        acc += wd[ky * 5 + kx] * gc[iy * WW + ix];
      }
    }
    As[r * 64 + c] = (_Float16)acc;       // A[m=pixel][k=channel]
  }
  for (int i = tid; i < (64 - HW) * 64; i += 256)
    As[HW * 64 + i] = (_Float16)0.f;      // zero pad rows 49..63
  __syncthreads();

  const int wave = tid >> 5, lane = tid & 31;
  const int half = lane >> 4, l15 = lane & 15;
  for (int tIdx = wave; tIdx < 16; tIdx += 8) {
    const int mi = tIdx >> 2, nj = tIdx & 3;
    const int row = mi * 16 + l15;
    const int col = nj * 16 + l15;
    v8f acc = {};
#pragma unroll
    for (int ks = 0; ks < 64; ks += 32) {
      const _Float16* ap = As + row * 64 + ks + half * 16;
      const v8h alo = *(const v8h*)ap;          // K = ks+kbA .. +7
      const v8h ahi = *(const v8h*)(ap + 8);    // K = ks+kbA+8 .. +15
      const v16h a  = join16(alo, ahi);
      const v16h b  = *(const v16h*)(pw + ((size_t)e * CPT + col) * CPT + ks + half * 16);
      acc = __builtin_amdgcn_wmma_f32_16x16x32_f16(false, a, false, b,
                                                   (short)0, acc, false, false);
    }
    const float bv = pw_b[e * CPT + col];
    float* db = dstfeat + (((size_t)n * P + dpt) * CPT + col) * HW;
#pragma unroll
    for (int v = 0; v < 8; ++v) {
      const int mr = mi * 16 + half * 8 + v;
      if (mr < HW) atomicAdd(&db[mr], acc[v] + bv);
    }
  }
}

// ---------------------------------------------------------------------------
extern "C" void kernel_launch(void* const* d_in, const int* in_sizes, int n_in,
                              void* d_out, int out_size, void* d_ws, size_t ws_size,
                              hipStream_t stream) {
  (void)in_sizes; (void)n_in; (void)out_size; (void)ws_size;
  const float* x       = (const float*)d_in[0];
  const float* conv0_w = (const float*)d_in[1];
  const float* conv0_b = (const float*)d_in[2];
  const float* gn0_s   = (const float*)d_in[3];
  const float* gn0_b   = (const float*)d_in[4];
  const float* convs_w = (const float*)d_in[5];
  const float* convs_b = (const float*)d_in[6];
  const float* gns_s   = (const float*)d_in[7];
  const float* gns_b   = (const float*)d_in[8];
  const float* fo_dw_w = (const float*)d_in[9];
  const float* fo_dw_b = (const float*)d_in[10];
  const float* fo_pw_w = (const float*)d_in[11];
  const float* fo_pw_b = (const float*)d_in[12];
  const float* so_dw_w = (const float*)d_in[13];
  const float* so_dw_b = (const float*)d_in[14];
  const float* so_pw_w = (const float*)d_in[15];
  const float* so_pw_b = (const float*)d_in[16];

  const size_t actElems = (size_t)NROI * COUT * HW;      // 14,450,688
  float*    b0   = (float*)d_ws;
  float*    b1   = b0 + actElems;
  _Float16* wp0  = (_Float16*)(b1 + actElems);
  _Float16* wpk  = wp0 + (size_t)COUT * 9 * DIMIN;
  _Float16* pwfo = wpk + (size_t)7 * COUT * 9 * COUT;
  _Float16* pwso = pwfo + (size_t)NE * CPT * CPT;

  // --- pack weights to f16 WMMA-friendly layouts ---
  {
    const size_t t0 = (size_t)COUT * DIMIN * 9;
    pack_conv_w<<<dim3((unsigned)((t0 + 255) / 256)), 256, 0, stream>>>(conv0_w, wp0, COUT, DIMIN);
    const size_t tk = (size_t)COUT * COUT * 9;
    for (int k = 0; k < 7; ++k)
      pack_conv_w<<<dim3((unsigned)((tk + 255) / 256)), 256, 0, stream>>>(
          convs_w + (size_t)k * tk, wpk + (size_t)k * COUT * 9 * COUT, COUT, COUT);
    const int tp = NE * CPT * CPT;
    cvt_f32_to_f16<<<dim3((tp + 255) / 256), 256, 0, stream>>>(fo_pw_w, pwfo, tp);
    cvt_f32_to_f16<<<dim3((tp + 255) / 256), 256, 0, stream>>>(so_pw_w, pwso, tp);
  }

  // --- conv stack ---
  conv3x3_gemm_wmma<<<dim3(NROI, COUT / 32), 256,
                      (size_t)DIMIN * 16 * 16 * sizeof(_Float16), stream>>>(
      x, wp0, conv0_b, b0, DIMIN, 14, 14, 2);
  gn_relu<<<dim3(NROI, GROUPS), 256, 0, stream>>>(b0, gn0_s, gn0_b);

  float* cur = b0;
  float* nxt = b1;
  for (int k = 0; k < 7; ++k) {
    conv3x3_gemm_wmma<<<dim3(NROI, COUT / 32), 256,
                        (size_t)COUT * 9 * 9 * sizeof(_Float16), stream>>>(
        cur, wpk + (size_t)k * COUT * 9 * COUT, convs_b + k * COUT, nxt, COUT, 7, 7, 1);
    gn_relu<<<dim3(NROI, GROUPS), 256, 0, stream>>>(nxt, gns_s + k * COUT, gns_b + k * COUT);
    float* t = cur; cur = nxt; nxt = t;
  }
  float* h  = cur;   // after 7 swaps: b1
  float* fo = nxt;   // b0, free for reuse

  const size_t actBytes = actElems * sizeof(float);
  float* outF = (float*)d_out;

  // output[0] = h ; fo = pts ; fo += msgs_fo(pts[SRC])
  hipMemcpyAsync(outF, h, actBytes, hipMemcpyDeviceToDevice, stream);
  hipMemcpyAsync(fo,   h, actBytes, hipMemcpyDeviceToDevice, stream);
  msg_dwpw_wmma<<<dim3(NROI, NE), 256, 0, stream>>>(h, fo, fo_dw_w, fo_dw_b, pwfo, fo_pw_b);

  // so = pts ; so += msgs_so(fo[SRC]) ; output[1] = so
  hipMemcpyAsync(outF + actElems, h, actBytes, hipMemcpyDeviceToDevice, stream);
  msg_dwpw_wmma<<<dim3(NROI, NE), 256, 0, stream>>>(fo, outF + actElems,
                                                    so_dw_w, so_dw_b, pwso, so_pw_b);
}
